// CausalSelfAttention_35390530519661
// MI455X (gfx1250) — matrix operand order repaired and checked by reading, not compiled, not run on hardware
//
#include <hip/hip_runtime.h>

typedef _Float16 v16h __attribute__((ext_vector_type(16)));
typedef _Float16 v8h  __attribute__((ext_vector_type(8)));
typedef float    v8f  __attribute__((ext_vector_type(8)));

#define BATCH 4
#define TSEQ  2048
#define CDIM  1024
#define NHEAD 16
#define HDIM  64
#define C3    3072

// ---------------------------------------------------------------------------
// Prep: fp32 -> f16 convert, and transposed convert for weights (W^T layout
// makes WMMA B-fragments contiguous 32B per-lane loads).
// ---------------------------------------------------------------------------
__global__ __launch_bounds__(256) void k_cvt_f16(const float* __restrict__ in,
                                                 _Float16* __restrict__ out, int n) {
  int i = blockIdx.x * 256 + threadIdx.x;
  if (i < n) out[i] = (_Float16)in[i];
}

// in: [K][N] row-major fp32  ->  out: [N][K] row-major f16
__global__ __launch_bounds__(256) void k_transpose_f16(const float* __restrict__ in,
                                                       _Float16* __restrict__ out,
                                                       int K, int N) {
  int i = blockIdx.x * 256 + threadIdx.x;
  if (i >= K * N) return;
  int k = i % K;
  int n = i / K;
  out[(size_t)n * K + k] = (_Float16)in[(size_t)k * N + n];
}

// ---------------------------------------------------------------------------
// GEMM: C[M,N] = A[M,K] * B[K,N], A f16 row-major, B given as BT[N][K] f16.
// One wave computes a 32x64 tile: 8 independent accumulators per k-step so
// dependent WMMAs are >= 8 apart (hides the 5-slot f16 WMMA RAW hazard), and
// operand traffic drops to 24 B/WMMA ((64B A + 128B B) / 8).
// ---------------------------------------------------------------------------
template <bool OUT_HALF>
__global__ __launch_bounds__(256) void k_gemm_wmma(const _Float16* __restrict__ A,
                                                   const _Float16* __restrict__ BT,
                                                   void* __restrict__ Cp,
                                                   int M, int N, int K) {
  const int lane = threadIdx.x & 31;
  const int gw   = blockIdx.x * 8 + (threadIdx.x >> 5);
  const int nt64 = N >> 6;
  const int m0   = (gw / nt64) << 5;   // 32-row tile
  const int n0   = (gw % nt64) << 6;   // 64-col tile
  if (m0 >= M) return;  // wave-uniform
  const int hi  = (lane >> 4) & 1;
  const int r16 = lane & 15;

  v8f acc[2][4] = {};
  const _Float16* arow0 = A + (size_t)(m0 + r16) * K;
  const _Float16* arow1 = A + (size_t)(m0 + 16 + r16) * K;

  for (int k0 = 0; k0 < K; k0 += 32) {
    __builtin_prefetch(arow0 + k0 + 256, 0, 1);
    // A-frags 16x32: lanes 0-15 hold K 0-15, lanes 16-31 hold K 16-31
    v16h a0 = *(const v16h*)(arow0 + k0 + hi * 16);
    v16h a1 = *(const v16h*)(arow1 + k0 + hi * 16);
#pragma unroll
    for (int nn = 0; nn < 4; ++nn) {
      // B-frag 32x16: lane = column n, contiguous K run (0-15 or 16-31) from BT row
      const _Float16* brow = BT + (size_t)(n0 + nn * 16 + r16) * K + k0 + hi * 16;
      v16h b = *(const v16h*)brow;
      acc[0][nn] = __builtin_amdgcn_wmma_f32_16x16x32_f16(false, a0, false, b,
                                                          (short)0, acc[0][nn], false, false);
      acc[1][nn] = __builtin_amdgcn_wmma_f32_16x16x32_f16(false, a1, false, b,
                                                          (short)0, acc[1][nn], false, false);
    }
  }
  // C layout: lane holds column n0+lane%16; VGPR r holds row m0 + r + 8*(lane>=16)
#pragma unroll
  for (int mm = 0; mm < 2; ++mm) {
#pragma unroll
    for (int nn = 0; nn < 4; ++nn) {
#pragma unroll
      for (int r = 0; r < 8; ++r) {
        size_t idx = (size_t)(m0 + mm * 16 + r + 8 * hi) * N + n0 + nn * 16 + r16;
        if (OUT_HALF) ((_Float16*)Cp)[idx] = (_Float16)acc[mm][nn][r];
        else          ((float*)Cp)[idx]    = acc[mm][nn][r];
      }
    }
  }
}

// ---------------------------------------------------------------------------
// RoPE + head split. qkv f16 [b*t][3072] -> Qh,Kh f16 [bh][t][64], V transposed
// Vt f16 [bh][64][t]. Folds 1/sqrt(HDIM) into Q.
// ---------------------------------------------------------------------------
__global__ __launch_bounds__(256) void k_rope_split(const _Float16* __restrict__ qkv,
                                                    _Float16* __restrict__ Qh,
                                                    _Float16* __restrict__ Kh,
                                                    _Float16* __restrict__ Vt) {
  int gid = blockIdx.x * 256 + threadIdx.x;     // [0, 4*2048*16*32)
  int i = gid & 31;                              // rotation pair index
  int h = (gid >> 5) & 15;
  int t = (gid >> 9) & 2047;
  int b = gid >> 20;
  size_t row = (size_t)(b * TSEQ + t) * C3;
  int col = h * HDIM + i;

  float inv = __powf(10000.0f, -(float)(2 * i) * (1.0f / (float)HDIM));
  float ang = inv * (float)t;
  float cs = __cosf(ang), sn = __sinf(ang);

  float ql = (float)qkv[row + col];
  float qh = (float)qkv[row + col + 32];
  float kl = (float)qkv[row + CDIM + col];
  float kh = (float)qkv[row + CDIM + col + 32];
  float vl = (float)qkv[row + 2 * CDIM + col];
  float vh = (float)qkv[row + 2 * CDIM + col + 32];

  size_t bh = (size_t)(b * NHEAD + h);
  size_t qk = (bh * TSEQ + t) * HDIM;
  Qh[qk + i]      = (_Float16)((ql * cs - qh * sn) * 0.125f);
  Qh[qk + i + 32] = (_Float16)((qh * cs + ql * sn) * 0.125f);
  Kh[qk + i]      = (_Float16)(kl * cs - kh * sn);
  Kh[qk + i + 32] = (_Float16)(kh * cs + kl * sn);
  Vt[(bh * HDIM + i) * TSEQ + t]        = (_Float16)vl;
  Vt[(bh * HDIM + i + 32) * TSEQ + t]   = (_Float16)vh;
}

// ---------------------------------------------------------------------------
// One 32-key flash block. MASKED only for the final (diagonal) block.
// Row sum is kept as a per-lane partial (lpart) -- the rescale factor is
// uniform across the 16 lanes of a row, so the cross-lane sum reduction is
// deferred to the end of the kernel. Only the row-max needs an in-loop
// reduction (P is cast to f16; a loose max would underflow whole rows).
// ---------------------------------------------------------------------------
template <bool MASKED>
__device__ __forceinline__ void flash_block(int k0, int q0, int hi, int r16,
                                            const _Float16* __restrict__ Kp,
                                            const _Float16* __restrict__ Vp,
                                            _Float16* myp,
                                            const v16h (&aq)[2],
                                            float (&mrow)[8], float (&lpart)[8],
                                            v8f (&o)[4]) {
  // S tiles (keys k0..k0+15 and k0+16..k0+31); s-major order interleaves the
  // two independent accumulators between dependent WMMAs.
  v8f st[2] = {};
#pragma unroll
  for (int s = 0; s < 2; ++s) {
#pragma unroll
    for (int tile = 0; tile < 2; ++tile) {
      const _Float16* kr = Kp + (size_t)(k0 + tile * 16 + r16) * HDIM + s * 32 + hi * 16;
      v16h kb = *(const v16h*)kr;
      st[tile] = __builtin_amdgcn_wmma_f32_16x16x32_f16(false, aq[s], false, kb,
                                                        (short)0, st[tile], false, false);
    }
  }
  // online softmax statistics (row = r + 8*hi, col = lane%16)
  float scale[8];
#pragma unroll
  for (int r = 0; r < 8; ++r) {
    float s0 = st[0][r];
    float s1 = st[1][r];
    if (MASKED) {
      int qrow = q0 + r + 8 * hi;
      s0 = (k0 + r16 > qrow)      ? -1e30f : s0;
      s1 = (k0 + 16 + r16 > qrow) ? -1e30f : s1;
    }
    float mx = fmaxf(s0, s1);
    mx = fmaxf(mx, __shfl_xor(mx, 1, 32));
    mx = fmaxf(mx, __shfl_xor(mx, 2, 32));
    mx = fmaxf(mx, __shfl_xor(mx, 4, 32));
    mx = fmaxf(mx, __shfl_xor(mx, 8, 32));
    float mnew = fmaxf(mrow[r], mx);
    float sc8  = __expf(mrow[r] - mnew);
    mrow[r] = mnew;
    scale[r] = sc8;
    float p0 = __expf(s0 - mnew);
    float p1 = __expf(s1 - mnew);
    st[0][r] = p0;
    st[1][r] = p1;
    lpart[r] = lpart[r] * sc8 + p0 + p1;   // per-lane partial; reduced at end
  }
#pragma unroll
  for (int nn = 0; nn < 4; ++nn)
#pragma unroll
    for (int r = 0; r < 8; ++r) o[nn][r] *= scale[r];

  // P (C-layout) -> LDS -> A-fragment relayout
#pragma unroll
  for (int tile = 0; tile < 2; ++tile)
#pragma unroll
    for (int r = 0; r < 8; ++r)
      myp[(r + 8 * hi) * 32 + tile * 16 + r16] = (_Float16)st[tile][r];
  asm volatile("s_wait_dscnt 0x0" ::: "memory");
  v16h pa = *(const v16h*)(myp + r16 * 32 + hi * 16);

  // O += P * V ; V^T layout makes B-frag a contiguous 32B load
#pragma unroll
  for (int nn = 0; nn < 4; ++nn) {
    const _Float16* vr = Vp + (size_t)(nn * 16 + r16) * TSEQ + k0 + hi * 16;
    v16h vb = *(const v16h*)vr;
    o[nn] = __builtin_amdgcn_wmma_f32_16x16x32_f16(false, pa, false, vb,
                                                   (short)0, o[nn], false, false);
  }
}

// ---------------------------------------------------------------------------
// Flash attention: one wave per (bh, 16-row q block). Unmasked main loop +
// peeled masked diagonal block (only the final 32-key block can be masked).
// ---------------------------------------------------------------------------
__global__ __launch_bounds__(256) void k_flash(const _Float16* __restrict__ Qh,
                                               const _Float16* __restrict__ Kh,
                                               const _Float16* __restrict__ Vt,
                                               _Float16* __restrict__ Y) {
  __shared__ _Float16 plds[8][16 * 32];
  const int lane = threadIdx.x & 31;
  const int wid  = threadIdx.x >> 5;
  const int gw   = blockIdx.x * 8 + wid;   // 0..8191
  const int bh   = gw >> 7;                // 0..63
  const int q0   = (gw & 127) << 4;        // 0..2032
  const int b    = bh >> 4, h = bh & 15;
  const int hi   = (lane >> 4) & 1;
  const int r16  = lane & 15;
  const _Float16* Qp = Qh + (size_t)bh * TSEQ * HDIM;
  const _Float16* Kp = Kh + (size_t)bh * TSEQ * HDIM;
  const _Float16* Vp = Vt + (size_t)bh * HDIM * TSEQ;
  _Float16* myp = &plds[wid][0];

  // Q A-fragments for hd 0-31 and 32-63
  v16h aq[2];
#pragma unroll
  for (int s = 0; s < 2; ++s) {
    const _Float16* qr = Qp + (size_t)(q0 + r16) * HDIM + s * 32 + hi * 16;
    aq[s] = *(const v16h*)qr;
  }

  float mrow[8], lpart[8];
#pragma unroll
  for (int r = 0; r < 8; ++r) { mrow[r] = -1e30f; lpart[r] = 0.0f; }
  v8f o[4] = {};

  // last 32-key block start (the only one that can intersect the diagonal)
  const int kLast = ((q0 + 15) >> 5) << 5;
  for (int k0 = 0; k0 < kLast; k0 += 32)
    flash_block<false>(k0, q0, hi, r16, Kp, Vp, myp, aq, mrow, lpart, o);
  flash_block<true>(kLast, q0, hi, r16, Kp, Vp, myp, aq, mrow, lpart, o);

  // deferred cross-lane row-sum reduction, then normalize and write y
#pragma unroll
  for (int r = 0; r < 8; ++r) {
    float rs = lpart[r];
    rs += __shfl_xor(rs, 1, 32);
    rs += __shfl_xor(rs, 2, 32);
    rs += __shfl_xor(rs, 4, 32);
    rs += __shfl_xor(rs, 8, 32);
    float invl = 1.0f / rs;
    size_t row = (size_t)(b * TSEQ + q0 + r + 8 * hi);
#pragma unroll
    for (int nn = 0; nn < 4; ++nn)
      Y[row * CDIM + h * HDIM + nn * 16 + r16] = (_Float16)(o[nn][r] * invl);
  }
}

// ---------------------------------------------------------------------------
extern "C" void kernel_launch(void* const* d_in, const int* in_sizes, int n_in,
                              void* d_out, int out_size, void* d_ws, size_t ws_size,
                              hipStream_t stream) {
  (void)in_sizes; (void)n_in; (void)out_size; (void)ws_size;
  const float* x      = (const float*)d_in[0];   // [4,2048,1024]
  const float* w_attn = (const float*)d_in[1];   // [1024,3072]
  const float* w_proj = (const float*)d_in[2];   // [1024,1024]
  float* out = (float*)d_out;                    // [4,2048,1024] fp32

  char* ws = (char*)d_ws;
  size_t off = 0;
  _Float16* xb  = (_Float16*)(ws + off); off += (size_t)BATCH * TSEQ * CDIM * 2;   // 16 MB
  _Float16* waT = (_Float16*)(ws + off); off += (size_t)C3 * CDIM * 2;             // 6 MB
  _Float16* wpT = (_Float16*)(ws + off); off += (size_t)CDIM * CDIM * 2;           // 2 MB
  _Float16* qkv = (_Float16*)(ws + off);
  _Float16* Yb  = qkv;  // reuse: qkv dead after rope-split, Yb written by flash
  off += (size_t)BATCH * TSEQ * C3 * 2;                                            // 48 MB
  _Float16* Qh = (_Float16*)(ws + off); off += (size_t)BATCH * NHEAD * TSEQ * HDIM * 2;
  _Float16* Kh = (_Float16*)(ws + off); off += (size_t)BATCH * NHEAD * TSEQ * HDIM * 2;
  _Float16* Vt = (_Float16*)(ws + off); off += (size_t)BATCH * NHEAD * TSEQ * HDIM * 2;

  const int M = BATCH * TSEQ;  // 8192

  // 1) convert x, transpose-convert weights
  k_cvt_f16<<<(M * CDIM) / 256, 256, 0, stream>>>(x, xb, M * CDIM);
  k_transpose_f16<<<(CDIM * C3) / 256, 256, 0, stream>>>(w_attn, waT, CDIM, C3);
  k_transpose_f16<<<(CDIM * CDIM) / 256, 256, 0, stream>>>(w_proj, wpT, CDIM, CDIM);

  // 2) qkv = x @ w_attn : waves = (8192/32)*(3072/64) = 12288 -> 1536 blocks
  k_gemm_wmma<true><<<1536, 256, 0, stream>>>(xb, waT, (void*)qkv, M, C3, CDIM);

  // 3) RoPE + head split (+ fold 1/sqrt(hd) into Q, transpose V)
  k_rope_split<<<(BATCH * TSEQ * NHEAD * 32) / 256, 256, 0, stream>>>(qkv, Qh, Kh, Vt);

  // 4) flash attention: 64 bh * 128 q-blocks = 8192 waves -> 1024 blocks
  k_flash<<<1024, 256, 0, stream>>>(Qh, Kh, Vt, Yb);

  // 5) out = y @ w_proj : waves = (8192/32)*(1024/64) = 4096 -> 512 blocks
  k_gemm_wmma<false><<<512, 256, 0, stream>>>(Yb, wpT, (void*)out, M, CDIM, CDIM);
}
